// GCRN_8418135900356
// MI455X (gfx1250) — matrix operand order repaired and checked by reading, not compiled, not run on hardware
//
#include <hip/hip_runtime.h>
#include <cstdint>
#include <cstddef>

// =====================================================================
// AGCRN forward for MI455X (gfx1250, wave32, WMMA bf16 16x16x32).
//
//  * A = softmax(relu(E E^T))  : time-invariant, built once (bf16)
//  * Wn = sum_d E[n,d]*Wp[d]   : time-invariant, bf16, stored K-major
//    ("fragment-friendly") so WMMA B-fragments are contiguous 32B loads.
//  * Graph conv G = A @ XH     : 1024x1024xK GEMM, bf16 WMMA, 128x128
//    block tile, double-buffered LDS with register prefetch, B tile
//    transposed in LDS so all fragment reads are ds_load_b128.
//  * Gate: out[n] = [XH_n|G_n] @ Wn + b_n, per-node 64 x 2C x Og WMMA.
//  * GRU elementwise + static (adj@xt) path in fp32.
// Workspace requirement: ~390 MB.
// =====================================================================

typedef __attribute__((ext_vector_type(16))) __bf16 v16bf;
typedef __attribute__((ext_vector_type(8)))  __bf16 v8bf;
typedef __attribute__((ext_vector_type(8)))  float  v8f;

#define NN 1024
#define BB 64
#define TT 12
#define DD 10
#define CO 64

static __device__ __forceinline__ v16bf cat8(v8bf lo, v8bf hi) {
  return __builtin_shufflevector(lo, hi, 0, 1, 2, 3, 4, 5, 6, 7,
                                 8, 9, 10, 11, 12, 13, 14, 15);
}

// ---------------------------------------------------------------------
// A = softmax(relu(E E^T)) row-wise, written as bf16. One block per row.
__global__ __launch_bounds__(256)
void k_build_A(const float* __restrict__ E, __bf16* __restrict__ Abf) {
  int n = blockIdx.x;
  int t = threadIdx.x;
  __shared__ float red[256];
  float en[DD];
#pragma unroll
  for (int d = 0; d < DD; ++d) en[d] = E[n * DD + d];
  float v[4];
#pragma unroll
  for (int i = 0; i < 4; ++i) {
    int m = t + i * 256;
    float acc = 0.f;
#pragma unroll
    for (int d = 0; d < DD; ++d) acc += en[d] * E[m * DD + d];
    v[i] = acc > 0.f ? acc : 0.f;
  }
  float mx = fmaxf(fmaxf(v[0], v[1]), fmaxf(v[2], v[3]));
  red[t] = mx; __syncthreads();
  for (int s = 128; s > 0; s >>= 1) { if (t < s) red[t] = fmaxf(red[t], red[t + s]); __syncthreads(); }
  mx = red[0]; __syncthreads();
  float sum = 0.f;
#pragma unroll
  for (int i = 0; i < 4; ++i) { v[i] = __expf(v[i] - mx); sum += v[i]; }
  red[t] = sum; __syncthreads();
  for (int s = 128; s > 0; s >>= 1) { if (t < s) red[t] += red[t + s]; __syncthreads(); }
  float inv = 1.f / red[0];
#pragma unroll
  for (int i = 0; i < 4; ++i) Abf[(size_t)n * NN + t + i * 256] = (__bf16)(v[i] * inv);
}

// ---------------------------------------------------------------------
__global__ void k_f32_to_bf16(const float* __restrict__ src, __bf16* __restrict__ dst, int n) {
  int i = blockIdx.x * blockDim.x + threadIdx.x;
  if (i < n) dst[i] = (__bf16)src[i];
}

// ---------------------------------------------------------------------
// K-major swizzled per-node weights: Wn[n][o][k] = sum_d E[n,d]*Wp[d,kk,i,o]
// k in [0,C)->support0, [C,2C)->support1, [2C,Kp) zero padding.
__global__ void k_build_Wn(const float* __restrict__ E, const float* __restrict__ Wp,
                           __bf16* __restrict__ Wn, int C, int Kp, int Og) {
  long long idx = (long long)blockIdx.x * blockDim.x + threadIdx.x;
  long long total = (long long)NN * Og * Kp;
  if (idx >= total) return;
  int k = (int)(idx % Kp);
  long long r = idx / Kp;
  int o = (int)(r % Og);
  int n = (int)(r / Og);
  float acc = 0.f;
  if (k < 2 * C) {
    int kk = (k < C) ? 0 : 1;
    int i  = (k < C) ? k : k - C;
#pragma unroll
    for (int d = 0; d < DD; ++d)
      acc += E[n * DD + d] * Wp[(((long long)d * 2 + kk) * C + i) * Og + o];
  }
  Wn[idx] = (__bf16)acc;
}

// bias_n[o] = sum_d E[n,d] * bp[d,o]
__global__ void k_build_bias(const float* __restrict__ E, const float* __restrict__ bp,
                             float* __restrict__ bias, int Og) {
  int idx = blockIdx.x * blockDim.x + threadIdx.x;
  if (idx >= NN * Og) return;
  int o = idx % Og, n = idx / Og;
  float acc = 0.f;
#pragma unroll
  for (int d = 0; d < DD; ++d) acc += E[n * DD + d] * bp[d * Og + o];
  bias[idx] = acc;
}

// ---------------------------------------------------------------------
__global__ void k_init_h(const float* __restrict__ ist, float* __restrict__ h, int layer) {
  int idx = blockIdx.x * blockDim.x + threadIdx.x;
  if (idx >= NN * BB * CO) return;
  int o = idx % CO; int r = idx / CO; int b = r % BB; int n = r / BB;
  h[idx] = ist[(((size_t)layer * BB + b) * NN + n) * CO + o];
}

// ---------------------------------------------------------------------
// XH[n][b][c] (bf16): c<Ci from x (layer0 fp32 input or layer0-output
// bf16 sequence), c>=Ci from h (optionally z*h with z = zr[...,0:64]).
__global__ void k_build_xh(const float* __restrict__ x0, const __bf16* __restrict__ xseq,
                           const float* __restrict__ h, const float* __restrict__ zr,
                           __bf16* __restrict__ XH, int t, int Ci, int C) {
  int idx = blockIdx.x * blockDim.x + threadIdx.x;
  if (idx >= NN * BB * C) return;
  int c = idx % C; int r = idx / C; int b = r % BB; int n = r / BB;
  float v;
  if (c < Ci) {
    if (x0) v = x0[(((size_t)b * TT + t) * NN + n) * Ci + c];
    else    v = (float)xseq[((size_t)t * NN + n) * (BB * CO) + b * CO + c];
  } else {
    int o = c - Ci;
    v = h[((size_t)n * BB + b) * CO + o];
    if (zr) v *= zr[((size_t)n * BB + b) * 128 + o];   // z gate (first half)
  }
  XH[((size_t)n * BB + b) * C + c] = (__bf16)v;
}

// xt0[n][b*2+c] = x[b,t,n,c]  (layer-0 static-path operand, bf16)
__global__ void k_build_xt0(const float* __restrict__ x, __bf16* __restrict__ xt0, int t) {
  int idx = blockIdx.x * blockDim.x + threadIdx.x;
  if (idx >= NN * BB * 2) return;
  int c = idx % 2; int r = idx / 2; int b = r % BB; int n = r / BB;
  xt0[idx] = (__bf16)x[(((size_t)b * TT + t) * NN + n) * 2 + c];
}

// ---------------------------------------------------------------------
// C[1024 x ncols] = Abf[1024x1024] * Bbf[1024 x ncols], bf16 WMMA f32 acc.
// Block tile 128x128, 8 waves: wave (wm,wn) -> rows wm*32 (2 m-tiles),
// cols wn*64 (4 n-tiles). Double-buffered LDS + register prefetch;
// one barrier per K-chunk. B tile stored K-major so fragment reads are
// contiguous ds_load_b128. Output dtype is a template param (no branchy
// epilogue), stores use pointer increments (no per-element 64-bit mul).
#define BSTR 40   // padded per-column stride (elements), 80B = 16B-aligned
template <bool OUT_BF16>
__global__ __launch_bounds__(256)
void k_gemm_t(const __bf16* __restrict__ Abf, const __bf16* __restrict__ Bbf,
              void* __restrict__ Cout, int ncols) {
  __shared__ alignas(16) __bf16 sA[2][128 * 32];   // row-major [m][k], 8KB each
  __shared__ alignas(16) __bf16 sB[2][128 * BSTR]; // K-major   [col][k], 10KB each
  int m0 = blockIdx.y * 128;
  int n0 = blockIdx.x * 128;
  int tid = threadIdx.x;
  int w = tid >> 5, lane = tid & 31;
  int wm = w & 3, wn = w >> 2;
  int lr = lane & 15, lh = lane >> 4;

  const uint32_t* gA = (const uint32_t*)Abf;       // 512 dwords per row
  const uint32_t* gB = (const uint32_t*)Bbf;
  int ldbd = ncols >> 1;

  uint32_t ra[8], u0[4], u1[4];
  auto fetch = [&](int kc) {
#pragma unroll
    for (int i = 0; i < 8; ++i) {                  // A tile: 128x32 = 2048 dwords
      int d = tid + i * 256;
      int row = d >> 4, cd = d & 15;
      ra[i] = gA[(size_t)(m0 + row) * 512 + (kc >> 1) + cd];
    }
#pragma unroll
    for (int i = 0; i < 4; ++i) {                  // B tile: 32x128
      int idx = tid + i * 256;
      int colpair = idx & 63;
      int kr = (idx >> 6) << 1;
      u0[i] = gB[(size_t)(kc + kr) * ldbd + (n0 >> 1) + colpair];
      u1[i] = gB[(size_t)(kc + kr + 1) * ldbd + (n0 >> 1) + colpair];
    }
  };
  auto stage = [&](int buf) {
    uint32_t* lA = (uint32_t*)sA[buf];
    uint32_t* lB = (uint32_t*)sB[buf];
#pragma unroll
    for (int i = 0; i < 8; ++i) lA[tid + i * 256] = ra[i];
#pragma unroll
    for (int i = 0; i < 4; ++i) {                  // 2x2 micro-transpose
      int idx = tid + i * 256;
      int colpair = idx & 63;
      int kr = (idx >> 6) << 1;
      uint32_t c0 = (u0[i] & 0xffffu) | (u1[i] << 16);
      uint32_t c1 = (u0[i] >> 16) | (u1[i] & 0xffff0000u);
      int col = colpair * 2;
      lB[col * (BSTR / 2) + (kr >> 1)] = c0;
      lB[(col + 1) * (BSTR / 2) + (kr >> 1)] = c1;
    }
  };

  v8f zero = {0.f, 0.f, 0.f, 0.f, 0.f, 0.f, 0.f, 0.f};
  v8f acc[2][4];
#pragma unroll
  for (int a = 0; a < 2; ++a)
#pragma unroll
    for (int c = 0; c < 4; ++c) acc[a][c] = zero;

  fetch(0);
  stage(0);
  int buf = 0;
  for (int kc = 0; kc < 1024; kc += 32) {
    __syncthreads();
    bool has_next = (kc + 32) < 1024;              // uniform
    if (has_next) fetch(kc + 32);

    const __bf16* cA = sA[buf];
    const __bf16* cB = sB[buf];
    v16bf fa[2];
#pragma unroll
    for (int a = 0; a < 2; ++a) {
      int m = wm * 32 + a * 16 + lr;
      v8bf lo = *(const v8bf*)&cA[m * 32 + 16 * lh];
      v8bf hi = *(const v8bf*)&cA[m * 32 + 8 + 16 * lh];
      fa[a] = cat8(lo, hi);
    }
#pragma unroll
    for (int c = 0; c < 4; ++c) {
      int col = wn * 64 + c * 16 + lr;
      v16bf fb = *(const v16bf*)&cB[col * BSTR + 16 * lh];
#pragma unroll
      for (int a = 0; a < 2; ++a)
        acc[a][c] = __builtin_amdgcn_wmma_f32_16x16x32_bf16(false, fa[a], false, fb,
                                                            (short)0, acc[a][c], false, false);
    }
    if (has_next) stage(buf ^ 1);
    buf ^= 1;
  }

#pragma unroll
  for (int a = 0; a < 2; ++a) {
#pragma unroll
    for (int c = 0; c < 4; ++c) {
      int col = n0 + wn * 64 + c * 16 + lr;
      int mbase = m0 + wm * 32 + a * 16 + 8 * lh;  // D: m = v + 8*half
      if (OUT_BF16) {
        __bf16* p = (__bf16*)Cout + (size_t)mbase * ncols + col;
#pragma unroll
        for (int vv = 0; vv < 8; ++vv) { *p = (__bf16)acc[a][c][vv]; p += ncols; }
      } else {
        float* p = (float*)Cout + (size_t)mbase * ncols + col;
#pragma unroll
        for (int vv = 0; vv < 8; ++vv) { *p = acc[a][c][vv]; p += ncols; }
      }
    }
  }
}

// ---------------------------------------------------------------------
// Per-node gate GEMM: out[n][b][o] = act( [XH_n | G_n][b,0:2C] @ Wn[n] + b_n )
// M=64 (batch), K=Kp, N=Og. One block per node. Wn is K-major [n][o][k]
// so B-fragments are contiguous 32-byte global loads (L2 resident).
// ACT: 0 = sigmoid, 1 = tanh (templated: no per-element branch).
template <int ACT>
__global__ __launch_bounds__(256)
void k_gate_t(const __bf16* __restrict__ XH, const __bf16* __restrict__ G,
              const __bf16* __restrict__ Wn, const float* __restrict__ bias,
              float* __restrict__ out, int C, int Kp, int Og) {
  int n = blockIdx.x;
  __shared__ alignas(16) __bf16 XG[BB * 256];      // [b][k], max Kp=256 -> 32KB
  int tid = threadIdx.x;
  // dword staging: C and Kp even, so bf16 pairs never straddle regions
  int kd = Kp >> 1;
  uint32_t* XGd = (uint32_t*)XG;
  int total2 = BB * kd;                            // multiple of 256
  for (int e = tid; e < total2; e += 256) {
    int c2 = (e % kd) * 2; int b = e / kd;
    uint32_t v = 0u;
    if (c2 < C)          v = *(const uint32_t*)&XH[((size_t)n * BB + b) * C + c2];
    else if (c2 < 2 * C) v = *(const uint32_t*)&G[((size_t)n * BB + b) * C + (c2 - C)];
    XGd[b * kd + (c2 >> 1)] = v;
  }
  __syncthreads();

  int w = tid >> 5, lane = tid & 31;
  int wm = w & 3, oh = w >> 2;                     // wave-uniform
  int lr = lane & 15, lh = lane >> 4;
  if (oh * 64 < Og) {
    v8f zero = {0.f, 0.f, 0.f, 0.f, 0.f, 0.f, 0.f, 0.f};
    v8f acc[4];
#pragma unroll
    for (int c = 0; c < 4; ++c) acc[c] = zero;
    const __bf16* wbase = Wn + (size_t)n * Og * Kp;
    int m = wm * 16 + lr;
    for (int kc = 0; kc < Kp; kc += 32) {
      v8bf lo = *(const v8bf*)&XG[m * Kp + kc + 16 * lh];
      v8bf hi = *(const v8bf*)&XG[m * Kp + kc + 8 + 16 * lh];
      v16bf fa = cat8(lo, hi);
#pragma unroll
      for (int c = 0; c < 4; ++c) {
        int o = oh * 64 + c * 16 + lr;
        v16bf fb = *(const v16bf*)(wbase + (size_t)o * Kp + kc + 16 * lh);
        acc[c] = __builtin_amdgcn_wmma_f32_16x16x32_bf16(false, fa, false, fb,
                                                         (short)0, acc[c], false, false);
      }
    }
#pragma unroll
    for (int c = 0; c < 4; ++c) {
      int o = oh * 64 + c * 16 + lr;
      float bv = bias[n * Og + o];
      float* p = out + ((size_t)n * BB + wm * 16 + 8 * lh) * Og + o;
#pragma unroll
      for (int vv = 0; vv < 8; ++vv) {
        float val = acc[c][vv] + bv;
        val = (ACT == 0) ? (1.f / (1.f + __expf(-val))) : tanhf(val);
        *p = val;
        p += Og;
      }
    }
  }
}

// ---------------------------------------------------------------------
__global__ void k_h_update(float* __restrict__ h, const float* __restrict__ zr,
                           const float* __restrict__ hc) {
  int idx = blockIdx.x * blockDim.x + threadIdx.x;
  if (idx >= NN * BB * CO) return;
  int o = idx % CO; int r2 = idx / CO; int b = r2 % BB; int n = r2 / BB;
  float r = zr[((size_t)n * BB + b) * 128 + 64 + o];
  h[idx] = r * h[idx] + (1.f - r) * hc[idx];
}

// layer-0 final: seq0[t][n][b*64+o] = k*h + (1-k)*(static@Wl + bl), bf16
__global__ void k_final0(const float* __restrict__ h, const float* __restrict__ st,
                         const float* __restrict__ Wl, const float* __restrict__ bl,
                         const float* __restrict__ ratio, __bf16* __restrict__ seq0, int t) {
  int idx = blockIdx.x * blockDim.x + threadIdx.x;
  if (idx >= NN * BB * CO) return;
  int o = idx % CO; int r = idx / CO; int b = r % BB; int n = r / BB;
  float kk = 0.5f + 0.5f * ratio[0];
  float s = bl[o]
          + st[(size_t)n * 128 + b * 2 + 0] * Wl[0 * CO + o]
          + st[(size_t)n * 128 + b * 2 + 1] * Wl[1 * CO + o];
  float f = kk * h[idx] + (1.f - kk) * s;
  seq0[((size_t)t * NN + n) * (BB * CO) + b * CO + o] = (__bf16)f;
}

// layer-1 final: d_out[b,t,n,o] = k*h + (1-k)*static
__global__ void k_final1(const float* __restrict__ h, const float* __restrict__ st,
                         const float* __restrict__ ratio, float* __restrict__ out, int t) {
  int idx = blockIdx.x * blockDim.x + threadIdx.x;
  if (idx >= NN * BB * CO) return;
  int o = idx % CO; int r = idx / CO; int b = r % BB; int n = r / BB;
  float kk = 0.5f + 0.5f * ratio[0];
  float f = kk * h[idx] + (1.f - kk) * st[idx];    // st is [N][B*64], == idx layout
  out[(((size_t)b * TT + t) * NN + n) * CO + o] = f;
}

// hidden[l,b,n,o] = h[n,b,o]
__global__ void k_write_hidden(const float* __restrict__ h, float* __restrict__ hid, int layer) {
  int idx = blockIdx.x * blockDim.x + threadIdx.x;
  if (idx >= NN * BB * CO) return;
  int o = idx % CO; int r = idx / CO; int b = r % BB; int n = r / BB;
  hid[(((size_t)layer * BB + b) * NN + n) * CO + o] = h[idx];
}

// =====================================================================
extern "C" void kernel_launch(void* const* d_in, const int* in_sizes, int n_in,
                              void* d_out, int out_size, void* d_ws, size_t ws_size,
                              hipStream_t stream) {
  (void)in_sizes; (void)n_in; (void)out_size; (void)ws_size;
  const float* x     = (const float*)d_in[0];
  const float* ist   = (const float*)d_in[1];
  const float* E     = (const float*)d_in[2];
  const float* adj   = (const float*)d_in[3];
  const float* ratio = (const float*)d_in[4];
  const float* Wg0 = (const float*)d_in[5];  const float* bg0 = (const float*)d_in[6];
  const float* Wu0 = (const float*)d_in[7];  const float* bu0 = (const float*)d_in[8];
  const float* Wg1 = (const float*)d_in[9];  const float* bg1 = (const float*)d_in[10];
  const float* Wu1 = (const float*)d_in[11]; const float* bu1 = (const float*)d_in[12];
  const float* Wl  = (const float*)d_in[13]; const float* bl  = (const float*)d_in[14];

  float* out     = (float*)d_out;
  float* out_hid = out + (size_t)BB * TT * NN * CO;

  // ---- workspace carve-up (256B aligned), total ~390MB ----
  char*  ws  = (char*)d_ws;
  size_t off = 0;
  auto alloc = [&](size_t bytes) -> char* {
    char* p = ws + off;
    off += (bytes + 255) & ~(size_t)255;
    return p;
  };
  __bf16* Abf   = (__bf16*)alloc((size_t)NN * NN * 2);
  __bf16* adjbf = (__bf16*)alloc((size_t)NN * NN * 2);
  __bf16* Wg0n  = (__bf16*)alloc((size_t)NN * 160 * 128 * 2);
  __bf16* Wu0n  = (__bf16*)alloc((size_t)NN * 160 * 64 * 2);
  __bf16* Wg1n  = (__bf16*)alloc((size_t)NN * 256 * 128 * 2);
  __bf16* Wu1n  = (__bf16*)alloc((size_t)NN * 256 * 64 * 2);
  float*  bg0n  = (float*)alloc((size_t)NN * 128 * 4);
  float*  bu0n  = (float*)alloc((size_t)NN * 64 * 4);
  float*  bg1n  = (float*)alloc((size_t)NN * 128 * 4);
  float*  bu1n  = (float*)alloc((size_t)NN * 64 * 4);
  __bf16* seq0  = (__bf16*)alloc((size_t)TT * NN * BB * CO * 2);
  __bf16* XH    = (__bf16*)alloc((size_t)NN * BB * 128 * 2);
  __bf16* Gb    = (__bf16*)alloc((size_t)NN * BB * 128 * 2);
  __bf16* xt0   = (__bf16*)alloc((size_t)NN * BB * 2 * 2);
  float*  zr    = (float*)alloc((size_t)NN * BB * 128 * 4);
  float*  hc    = (float*)alloc((size_t)NN * BB * 64 * 4);
  float*  h     = (float*)alloc((size_t)NN * BB * 64 * 4);
  float*  st    = (float*)alloc((size_t)NN * BB * 64 * 4);

  const int EW  = NN * BB * CO;         // 4,194,304 elementwise size
  const int EWB = EW / 256;             // 16384 blocks

  // ---- time-invariant precompute ----
  k_build_A<<<NN, 256, 0, stream>>>(E, Abf);
  k_f32_to_bf16<<<(NN * NN + 255) / 256, 256, 0, stream>>>(adj, adjbf, NN * NN);
  k_build_Wn<<<(int)(((long long)NN * 160 * 128 + 255) / 256), 256, 0, stream>>>(E, Wg0, Wg0n, 66, 160, 128);
  k_build_Wn<<<(int)(((long long)NN * 160 * 64 + 255) / 256), 256, 0, stream>>>(E, Wu0, Wu0n, 66, 160, 64);
  k_build_Wn<<<(int)(((long long)NN * 256 * 128 + 255) / 256), 256, 0, stream>>>(E, Wg1, Wg1n, 128, 256, 128);
  k_build_Wn<<<(int)(((long long)NN * 256 * 64 + 255) / 256), 256, 0, stream>>>(E, Wu1, Wu1n, 128, 256, 64);
  k_build_bias<<<(NN * 128 + 255) / 256, 256, 0, stream>>>(E, bg0, bg0n, 128);
  k_build_bias<<<(NN * 64 + 255) / 256, 256, 0, stream>>>(E, bu0, bu0n, 64);
  k_build_bias<<<(NN * 128 + 255) / 256, 256, 0, stream>>>(E, bg1, bg1n, 128);
  k_build_bias<<<(NN * 64 + 255) / 256, 256, 0, stream>>>(E, bu1, bu1n, 64);

  // ---- layer 0 (Ci=2, C=66, Kp=160, ncols = 64*66 = 4224) ----
  k_init_h<<<EWB, 256, 0, stream>>>(ist, h, 0);
  for (int t = 0; t < TT; ++t) {
    k_build_xh<<<(NN * BB * 66) / 256, 256, 0, stream>>>(x, nullptr, h, nullptr, XH, t, 2, 66);
    k_gemm_t<true><<<dim3(33, 8), 256, 0, stream>>>(Abf, XH, Gb, 4224);
    k_gate_t<0><<<NN, 256, 0, stream>>>(XH, Gb, Wg0n, bg0n, zr, 66, 160, 128);
    k_build_xh<<<(NN * BB * 66) / 256, 256, 0, stream>>>(x, nullptr, h, zr, XH, t, 2, 66);
    k_gemm_t<true><<<dim3(33, 8), 256, 0, stream>>>(Abf, XH, Gb, 4224);
    k_gate_t<1><<<NN, 256, 0, stream>>>(XH, Gb, Wu0n, bu0n, hc, 66, 160, 64);
    k_h_update<<<EWB, 256, 0, stream>>>(h, zr, hc);
    k_build_xt0<<<(NN * BB * 2 + 255) / 256, 256, 0, stream>>>(x, xt0, t);
    k_gemm_t<false><<<dim3(1, 8), 256, 0, stream>>>(adjbf, xt0, st, 128);
    k_final0<<<EWB, 256, 0, stream>>>(h, st, Wl, bl, ratio, seq0, t);
  }
  k_write_hidden<<<EWB, 256, 0, stream>>>(h, out_hid, 0);

  // ---- layer 1 (Ci=64, C=128, Kp=256, ncols = 64*128 = 8192) ----
  k_init_h<<<EWB, 256, 0, stream>>>(ist, h, 1);
  for (int t = 0; t < TT; ++t) {
    k_build_xh<<<(NN * BB * 128) / 256, 256, 0, stream>>>(nullptr, seq0, h, nullptr, XH, t, 64, 128);
    k_gemm_t<true><<<dim3(64, 8), 256, 0, stream>>>(Abf, XH, Gb, 8192);
    k_gate_t<0><<<NN, 256, 0, stream>>>(XH, Gb, Wg1n, bg1n, zr, 128, 256, 128);
    k_build_xh<<<(NN * BB * 128) / 256, 256, 0, stream>>>(nullptr, seq0, h, zr, XH, t, 64, 128);
    k_gemm_t<true><<<dim3(64, 8), 256, 0, stream>>>(Abf, XH, Gb, 8192);
    k_gate_t<1><<<NN, 256, 0, stream>>>(XH, Gb, Wu1n, bu1n, hc, 128, 256, 64);
    k_h_update<<<EWB, 256, 0, stream>>>(h, zr, hc);
    k_gemm_t<false><<<dim3(32, 8), 256, 0, stream>>>(adjbf, seq0 + (size_t)t * NN * BB * CO, st, 4096);
    k_final1<<<EWB, 256, 0, stream>>>(h, st, ratio, out, t);
  }
  k_write_hidden<<<EWB, 256, 0, stream>>>(h, out_hid, 1);
}
